// AutoregressivePredictor_48919677502031
// MI455X (gfx1250) — matrix operand order repaired and checked by reading, not compiled, not run on hardware
//
#include <hip/hip_runtime.h>
#include <hip/hip_bf16.h>

#define BB 4
#define SS 2048
#define DD 1024
#define HH 16
#define HDM 64
#define FFD 4096
#define MMR (BB*SS)   // 8192 rows of activations

typedef __bf16        v16bf __attribute__((ext_vector_type(16)));
typedef float         v8f   __attribute__((ext_vector_type(8)));
typedef unsigned int  v4u   __attribute__((ext_vector_type(4)));

union FragU { v16bf v; v4u q[2]; };

__device__ __forceinline__ unsigned short f2bf(float f) {
    unsigned int u = __builtin_bit_cast(unsigned int, f);
    u += 0x7FFFu + ((u >> 16) & 1u);            // round-to-nearest-even
    return (unsigned short)(u >> 16);
}
__device__ __forceinline__ float bf2f(unsigned short h) {
    unsigned int u = ((unsigned int)h) << 16;
    return __builtin_bit_cast(float, u);
}
__device__ __forceinline__ v8f vzero8() {
    v8f z = {0.f, 0.f, 0.f, 0.f, 0.f, 0.f, 0.f, 0.f};
    return z;
}

// ---------------- fp32 -> bf16 conversion (weights / activations) ----------------
__global__ __launch_bounds__(256) void k_convert_bf16(const float* __restrict__ src,
                                                      unsigned short* __restrict__ dst, int n) {
    int i = blockIdx.x * 256 + threadIdx.x;
    if (i < n) dst[i] = f2bf(src[i]);
}

// ---------------- RMSNorm (fp32 in, bf16 out) ----------------
__global__ __launch_bounds__(256) void k_rmsnorm_bf16(const float* __restrict__ x,
                                                      const float* __restrict__ w,
                                                      unsigned short* __restrict__ out, int dim) {
    const int row = blockIdx.x;
    const int tid = threadIdx.x;
    const float* xr = x + (size_t)row * dim;
    float ss = 0.f;
    for (int i = tid; i < dim; i += 256) { float v = xr[i]; ss += v * v; }
#pragma unroll
    for (int o = 16; o >= 1; o >>= 1) ss += __shfl_xor(ss, o, 32);
    __shared__ float red[8];
    const int wv = tid >> 5, ln = tid & 31;
    if (ln == 0) red[wv] = ss;
    __syncthreads();
    float tot = 0.f;
#pragma unroll
    for (int i = 0; i < 8; i++) tot += red[i];
    const float sc = rsqrtf(tot / (float)dim + 1e-6f);
    unsigned short* orow = out + (size_t)row * dim;
    for (int i = tid; i < dim; i += 256) orow[i] = f2bf(xr[i] * sc * w[i]);
}

// ---------------- bf16 GEMM: out[M,N] = A[M,K] @ W[N,K]^T (+residual)(+bias) ----------------
// Block: 256 threads = 8 waves (2 M x 4 N), block tile 64x256, wave tile 32x64.
// Fragment layouts per CDNA5 ISA 7.12.2 (wave32):
//   A 16x32 bf16: lane l holds row M=l%16; K = k0 + (l>=16?8:0)+{0..7} and k0+16+(l>=16?8:0)+{0..7}
//   B 32x16 bf16: lane l holds col N=l%16; K = k0 + (l>=16?8:0)+{0..7} and k0+16+(l>=16?8:0)+{0..7}
//   C 16x16 f32 : VGPR r -> row M=r+8*(l/16), col N=l%16
__global__ __launch_bounds__(256) void k_gemm_bf16(
    const unsigned short* __restrict__ A,
    const unsigned short* __restrict__ W,
    int N, int K,
    const float* __restrict__ residual,
    const float* __restrict__ bias,
    float* __restrict__ outF,
    unsigned short* __restrict__ outB) {
    const int tid  = threadIdx.x;
    const int wv   = tid >> 5, ln = tid & 31;
    const int wm   = wv >> 2,  wn = wv & 3;
    const int hl   = ln >> 4,  l16 = ln & 15;
    const int bm   = blockIdx.x * 64  + wm * 32;
    const int bn   = blockIdx.y * 256 + wn * 64;

    v8f acc[2][4];
#pragma unroll
    for (int mt = 0; mt < 2; mt++)
#pragma unroll
        for (int nt = 0; nt < 4; nt++) acc[mt][nt] = vzero8();

    const unsigned short* arow[2];
#pragma unroll
    for (int mt = 0; mt < 2; mt++)
        arow[mt] = A + (size_t)(bm + mt * 16 + l16) * K + hl * 8;
    const unsigned short* brow[4];
#pragma unroll
    for (int nt = 0; nt < 4; nt++)
        brow[nt] = W + (size_t)(bn + nt * 16 + l16) * K + hl * 8;

    for (int k0 = 0; k0 < K; k0 += 32) {
        FragU a[2], b[4];
#pragma unroll
        for (int mt = 0; mt < 2; mt++) {
            a[mt].q[0] = *(const v4u*)(arow[mt] + k0);
            a[mt].q[1] = *(const v4u*)(arow[mt] + k0 + 16);
        }
#pragma unroll
        for (int nt = 0; nt < 4; nt++) {
            b[nt].q[0] = *(const v4u*)(brow[nt] + k0);
            b[nt].q[1] = *(const v4u*)(brow[nt] + k0 + 16);
        }
        if (k0 + 64 < K) {                        // global_prefetch_b8 for next chunk
            __builtin_prefetch(arow[0] + k0 + 64, 0, 1);
            __builtin_prefetch(brow[0] + k0 + 64, 0, 1);
        }
#pragma unroll
        for (int mt = 0; mt < 2; mt++)
#pragma unroll
            for (int nt = 0; nt < 4; nt++)
                acc[mt][nt] = __builtin_amdgcn_wmma_f32_16x16x32_bf16(
                    false, a[mt].v, false, b[nt].v, (short)0, acc[mt][nt], false, false);
    }

#pragma unroll
    for (int mt = 0; mt < 2; mt++) {
#pragma unroll
        for (int nt = 0; nt < 4; nt++) {
#pragma unroll
            for (int r = 0; r < 8; r++) {
                const int row = bm + mt * 16 + r + hl * 8;
                const int col = bn + nt * 16 + l16;
                float v = acc[mt][nt][r];
                if (bias)     v += bias[col];
                if (residual) v += residual[(size_t)row * N + col];
                if (outF) outF[(size_t)row * N + col] = v;
                if (outB) outB[(size_t)row * N + col] = f2bf(v);
            }
        }
    }
}

// ---------------- RoPE (HF rotate_half) + head-transpose + V transpose ----------------
// qlin/klin/vlin: [B,S,H*HD] bf16 -> qr/kr: [B,H,S,HD] bf16 (q pre-scaled by 1/sqrt(HD)),
// vt: [B,H,HD,S] bf16 (so P@V B-fragments are contiguous loads).
__global__ __launch_bounds__(256) void k_rope_transpose(
    const unsigned short* __restrict__ qlin,
    const unsigned short* __restrict__ klin,
    const unsigned short* __restrict__ vlin,
    unsigned short* __restrict__ qr,
    unsigned short* __restrict__ kr,
    unsigned short* __restrict__ vt) {
    int idx = blockIdx.x * 256 + threadIdx.x;     // B*S*H*32 threads
    const int i = idx & 31;                       // rotation pair index
    int t = idx >> 5;
    const int h = t & (HH - 1);
    t >>= 4;
    const int s = t & (SS - 1);
    const int b = t >> 11;

    const float inv = __powf(10000.0f, -(float)i * (1.0f / 32.0f));
    float sn, cs;
    __sincosf((float)s * inv, &sn, &cs);

    const size_t src = ((size_t)(b * SS + s)) * DD + h * HDM;
    const float q1 = bf2f(qlin[src + i]), q2 = bf2f(qlin[src + i + 32]);
    const float k1 = bf2f(klin[src + i]), k2 = bf2f(klin[src + i + 32]);

    const size_t dst = (((size_t)(b * HH + h)) * SS + s) * HDM;
    const float qs = 0.125f;                      // 1/sqrt(64) folded into q
    qr[dst + i]      = f2bf((q1 * cs - q2 * sn) * qs);
    qr[dst + i + 32] = f2bf((q2 * cs + q1 * sn) * qs);
    kr[dst + i]      = f2bf(k1 * cs - k2 * sn);
    kr[dst + i + 32] = f2bf(k2 * cs + k1 * sn);

    const size_t vb = ((size_t)(b * HH + h)) * HDM * SS;
    vt[vb + (size_t)i * SS + s]        = vlin[src + i];
    vt[vb + (size_t)(i + 32) * SS + s] = vlin[src + i + 32];
}

// ---------------- Flash attention (full, non-causal) ----------------
// Block = 128 threads (4 waves), 64 queries per block (16/wave). Online softmax.
// S = Q@K^T via WMMA (K contiguous over HD); P re-layout via LDS; ctx += P@V^T.
__global__ __launch_bounds__(128) void k_attention(
    const unsigned short* __restrict__ Q,   // [B,H,S,HD] (roped, scaled)
    const unsigned short* __restrict__ Kc,  // [B,H,S,HD]
    const unsigned short* __restrict__ VT,  // [B,H,HD,S]
    unsigned short* __restrict__ ctx) {     // [B,S,H*HD]
    const int bh = blockIdx.x;
    const int b = bh >> 4, h = bh & 15;
    const int q0 = blockIdx.y * 64;
    const int wv = threadIdx.x >> 5, ln = threadIdx.x & 31;
    const int hl = ln >> 4, l16 = ln & 15;

    const size_t kbase = (size_t)bh * SS * HDM;
    const size_t vbase = (size_t)bh * HDM * SS;

    // Q fragments for the two HD chunks of 32; kept in VGPRs for the whole kernel.
    const unsigned short* qrow = Q + kbase + (size_t)(q0 + wv * 16 + l16) * HDM + hl * 8;
    FragU aq[2];
#pragma unroll
    for (int dk = 0; dk < 2; dk++) {
        aq[dk].q[0] = *(const v4u*)(qrow + dk * 32);
        aq[dk].q[1] = *(const v4u*)(qrow + dk * 32 + 16);
    }

    v8f cacc[4];
#pragma unroll
    for (int nt = 0; nt < 4; nt++) cacc[nt] = vzero8();
    float mrow[8], rrow[8];
#pragma unroll
    for (int r = 0; r < 8; r++) { mrow[r] = -3.0e38f; rrow[r] = 0.f; }

    __shared__ __align__(16) unsigned short pbuf[4][16][32];   // per-wave P tile

    for (int kc = 0; kc < SS; kc += 32) {
        // ---- scores: 16(q) x 32(k) per wave ----
        v8f sacc[2];
        sacc[0] = vzero8(); sacc[1] = vzero8();
        const unsigned short* kr0 = Kc + kbase + (size_t)(kc + l16) * HDM + hl * 8;
        const unsigned short* kr1 = Kc + kbase + (size_t)(kc + 16 + l16) * HDM + hl * 8;
#pragma unroll
        for (int dk = 0; dk < 2; dk++) {
            FragU b0, b1;
            b0.q[0] = *(const v4u*)(kr0 + dk * 32);
            b0.q[1] = *(const v4u*)(kr0 + dk * 32 + 16);
            b1.q[0] = *(const v4u*)(kr1 + dk * 32);
            b1.q[1] = *(const v4u*)(kr1 + dk * 32 + 16);
            sacc[0] = __builtin_amdgcn_wmma_f32_16x16x32_bf16(
                false, aq[dk].v, false, b0.v, (short)0, sacc[0], false, false);
            sacc[1] = __builtin_amdgcn_wmma_f32_16x16x32_bf16(
                false, aq[dk].v, false, b1.v, (short)0, sacc[1], false, false);
        }

        // ---- online softmax (row stats across the 16 lanes of each half-wave) ----
#pragma unroll
        for (int r = 0; r < 8; r++) {
            float s0 = sacc[0][r], s1 = sacc[1][r];
            float mx = fmaxf(s0, s1);
#pragma unroll
            for (int o = 1; o < 16; o <<= 1) mx = fmaxf(mx, __shfl_xor(mx, o, 16));
            const float mnew = fmaxf(mrow[r], mx);
            const float corr = __expf(mrow[r] - mnew);
            const float p0 = __expf(s0 - mnew);
            const float p1 = __expf(s1 - mnew);
            float ps = p0 + p1;
#pragma unroll
            for (int o = 1; o < 16; o <<= 1) ps += __shfl_xor(ps, o, 16);
            rrow[r] = rrow[r] * corr + ps;
            mrow[r] = mnew;
#pragma unroll
            for (int nt = 0; nt < 4; nt++) cacc[nt][r] *= corr;
            // C-layout -> LDS (row M = r+8*hl, cols kc+l16 and kc+16+l16)
            pbuf[wv][r + hl * 8][l16]      = f2bf(p0);
            pbuf[wv][r + hl * 8][16 + l16] = f2bf(p1);
        }
        __syncthreads();

        // ---- P as A-fragment from LDS, then ctx += P @ V^T ----
        FragU ap;
        ap.q[0] = *(const v4u*)(&pbuf[wv][l16][hl * 8]);
        ap.q[1] = *(const v4u*)(&pbuf[wv][l16][16 + hl * 8]);
#pragma unroll
        for (int nt = 0; nt < 4; nt++) {
            const unsigned short* vr = VT + vbase + (size_t)(nt * 16 + l16) * SS + kc + hl * 8;
            FragU bv;
            bv.q[0] = *(const v4u*)(vr);
            bv.q[1] = *(const v4u*)(vr + 16);
            cacc[nt] = __builtin_amdgcn_wmma_f32_16x16x32_bf16(
                false, ap.v, false, bv.v, (short)0, cacc[nt], false, false);
        }
    }

    // ---- normalize and store ctx in [B,S,H*HD] layout (ready for Wo GEMM) ----
    float invr[8];
#pragma unroll
    for (int r = 0; r < 8; r++) invr[r] = 1.0f / rrow[r];
    const int qi0 = q0 + wv * 16 + hl * 8;
#pragma unroll
    for (int nt = 0; nt < 4; nt++) {
#pragma unroll
        for (int r = 0; r < 8; r++) {
            const int qi = qi0 + r;
            const int d  = nt * 16 + l16;
            ctx[((size_t)(b * SS + qi)) * DD + h * HDM + d] = f2bf(cacc[nt][r] * invr[r]);
        }
    }
}

// ---------------- SwiGLU: out = silu(gate) * up (bf16, in-place over gate) ----------------
__global__ __launch_bounds__(256) void k_swiglu(const unsigned short* __restrict__ g,
                                                const unsigned short* __restrict__ u,
                                                unsigned short* __restrict__ o, int n) {
    int i = blockIdx.x * 256 + threadIdx.x;
    if (i >= n) return;
    const float gv = bf2f(g[i]);
    const float uv = bf2f(u[i]);
    const float sig = 1.0f / (1.0f + __expf(-gv));
    o[i] = f2bf(gv * sig * uv);
}

// ================================ host side ================================
extern "C" void kernel_launch(void* const* d_in, const int* in_sizes, int n_in,
                              void* d_out, int out_size, void* d_ws, size_t ws_size,
                              hipStream_t stream) {
    (void)in_sizes; (void)n_in; (void)out_size; (void)ws_size;
    const float* x    = (const float*)d_in[0];
    const float* Wq   = (const float*)d_in[1];
    const float* Wk   = (const float*)d_in[2];
    const float* Wv   = (const float*)d_in[3];
    const float* Wo   = (const float*)d_in[4];
    const float* ln1  = (const float*)d_in[5];
    const float* ln2  = (const float*)d_in[6];
    const float* Wg   = (const float*)d_in[7];
    const float* Wu   = (const float*)d_in[8];
    const float* Wd   = (const float*)d_in[9];
    const float* Wout = (const float*)d_in[10];
    const float* bout = (const float*)d_in[11];

    char* ws = (char*)d_ws;
    const size_t MB = 1024ull * 1024ull;
    // Aliased workspace plan (lifetimes disjoint along the dataflow), peak 226 MB:
    unsigned short* h1   = (unsigned short*)(ws + 0 * MB);     // 16MB, dead after V GEMM
    unsigned short* qlin = (unsigned short*)(ws + 16 * MB);    // 16MB, dead after rope
    unsigned short* klin = (unsigned short*)(ws + 32 * MB);
    unsigned short* vlin = (unsigned short*)(ws + 48 * MB);
    unsigned short* qr   = (unsigned short*)(ws + 64 * MB);    // dead after attention
    unsigned short* kr   = (unsigned short*)(ws + 80 * MB);
    unsigned short* vt   = (unsigned short*)(ws + 96 * MB);
    unsigned short* ctx  = (unsigned short*)(ws + 0 * MB);     // reuses h1
    float*          x1   = (float*)(ws + 16 * MB);             // 32MB, reuses qlin+klin
    unsigned short* h2   = (unsigned short*)(ws + 48 * MB);    // reuses vlin
    unsigned short* gate = (unsigned short*)(ws + 64 * MB);    // 64MB, reuses qr/kr/vt
    unsigned short* up   = (unsigned short*)(ws + 128 * MB);   // 64MB
    unsigned short* x2b  = (unsigned short*)(ws + 0 * MB);     // reuses ctx
    unsigned short* wqb  = (unsigned short*)(ws + 192 * MB);   // bf16 weight shadows
    unsigned short* wkb  = (unsigned short*)(ws + 194 * MB);
    unsigned short* wvb  = (unsigned short*)(ws + 196 * MB);
    unsigned short* wob  = (unsigned short*)(ws + 198 * MB);
    unsigned short* wgb  = (unsigned short*)(ws + 200 * MB);   // 8MB
    unsigned short* wub  = (unsigned short*)(ws + 208 * MB);   // 8MB
    unsigned short* wdb  = (unsigned short*)(ws + 216 * MB);   // 8MB
    unsigned short* wotb = (unsigned short*)(ws + 224 * MB);   // 2MB

    auto cvt = [&](const float* s, unsigned short* dptr, int n) {
        k_convert_bf16<<<(n + 255) / 256, 256, 0, stream>>>(s, dptr, n);
    };
    cvt(Wq, wqb, DD * DD);  cvt(Wk, wkb, DD * DD);
    cvt(Wv, wvb, DD * DD);  cvt(Wo, wob, DD * DD);
    cvt(Wg, wgb, FFD * DD); cvt(Wu, wub, FFD * DD);
    cvt(Wd, wdb, DD * FFD); cvt(Wout, wotb, DD * DD);

    // attention block
    k_rmsnorm_bf16<<<MMR, 256, 0, stream>>>(x, ln1, h1, DD);
    dim3 g1(MMR / 64, DD / 256);
    k_gemm_bf16<<<g1, 256, 0, stream>>>(h1, wqb, DD, DD, nullptr, nullptr, nullptr, qlin);
    k_gemm_bf16<<<g1, 256, 0, stream>>>(h1, wkb, DD, DD, nullptr, nullptr, nullptr, klin);
    k_gemm_bf16<<<g1, 256, 0, stream>>>(h1, wvb, DD, DD, nullptr, nullptr, nullptr, vlin);

    const int nrope = BB * SS * HH * 32;
    k_rope_transpose<<<nrope / 256, 256, 0, stream>>>(qlin, klin, vlin, qr, kr, vt);

    dim3 ga(BB * HH, SS / 64);
    k_attention<<<ga, 128, 0, stream>>>(qr, kr, vt, ctx);

    k_gemm_bf16<<<g1, 256, 0, stream>>>(ctx, wob, DD, DD, x, nullptr, x1, nullptr);

    // MLP block
    k_rmsnorm_bf16<<<MMR, 256, 0, stream>>>(x1, ln2, h2, DD);
    dim3 g2(MMR / 64, FFD / 256);
    k_gemm_bf16<<<g2, 256, 0, stream>>>(h2, wgb, FFD, DD, nullptr, nullptr, nullptr, gate);
    k_gemm_bf16<<<g2, 256, 0, stream>>>(h2, wub, FFD, DD, nullptr, nullptr, nullptr, up);

    const int nsw = MMR * FFD;
    k_swiglu<<<nsw / 256, 256, 0, stream>>>(gate, up, gate, nsw);

    k_gemm_bf16<<<g1, 256, 0, stream>>>(gate, wdb, DD, FFD, x1, nullptr, nullptr, x2b);

    // output head (writes fp32 logits straight into d_out)
    k_gemm_bf16<<<g1, 256, 0, stream>>>(x2b, wotb, DD, DD, nullptr, bout, (float*)d_out, nullptr);
}
